// _AttentionSelf_62234076119178
// MI455X (gfx1250) — compile-verified, matrix operand order checked by reading, not run on hardware
//
#include <hip/hip_runtime.h>

// ---------------------------------------------------------------------------
// Problem constants (match reference)
// ---------------------------------------------------------------------------
#define D_MODEL 2048
#define N_HEADS 16
#define D_KEY   128
#define BATCH   2
#define SEQ     2048
#define MTOT    (BATCH * SEQ)   // 4096 rows for the projection GEMMs

// ---------------------------------------------------------------------------
// CDNA5 WMMA types (wave32; v16bf A/B operands, v8f C/D accumulator)
// ---------------------------------------------------------------------------
typedef __attribute__((ext_vector_type(16))) __bf16 v16bf;
typedef __attribute__((ext_vector_type(8)))  __bf16 v8bf;
typedef __attribute__((ext_vector_type(8)))  float  v8f;

union Frag16 { v16bf v; v8bf h[2]; };

static __device__ __forceinline__ __bf16 f2bf(float f) {
  // round-to-nearest-even fp32 -> bf16
  unsigned u = __builtin_bit_cast(unsigned, f);
  unsigned r = u + 0x7FFFu + ((u >> 16) & 1u);
  unsigned short hv = (unsigned short)(r >> 16);
  return __builtin_bit_cast(__bf16, hv);
}

static __device__ __forceinline__ v8f wmma_bf16(v16bf a, v16bf b, v8f c) {
  // D = A(16x32) x B(32x16) + C(16x16 f32)
  return __builtin_amdgcn_wmma_f32_16x16x32_bf16(
      /*neg_a=*/false, a, /*neg_b=*/false, b,
      /*c_mod=*/(short)0, c, /*reuse_a=*/false, /*reuse_b=*/false);
}

// gfx1250 async global->LDS copy (ASYNCcnt-tracked). The LDS destination is
// the per-lane low 32 bits of the generic pointer; global addr in VGPR pair.
static __device__ __forceinline__ void async_ld_b128(void* lds, const void* g) {
  unsigned loff = (unsigned)(unsigned long long)lds;
  unsigned long long ga = (unsigned long long)g;
  asm volatile("global_load_async_to_lds_b128 %0, %1, off"
               :
               : "v"(loff), "v"(ga)
               : "memory");
}
static __device__ __forceinline__ void wait_async0() {
  asm volatile("s_wait_asynccnt 0" ::: "memory");
}

// 16-bit operand layout (ISA 7.12.2): lane L holds row (L%16); K chunks at
// kb=(L/16)*8 -> elements [kb..kb+7] and [16+kb..16+kb+7].
static __device__ __forceinline__ v16bf load_frag(const __bf16* rowbase,
                                                  int stride, int lane) {
  int row = lane & 15;
  int kb  = (lane >> 4) << 3;
  const __bf16* p = rowbase + row * stride;
  Frag16 f;
  f.h[0] = *(const v8bf*)(p + kb);
  f.h[1] = *(const v8bf*)(p + 16 + kb);
  return f.v;
}

// ---------------------------------------------------------------------------
// fp32 -> bf16 conversion (states + weights, done once per launch)
// ---------------------------------------------------------------------------
__global__ void cvt_f32_bf16(const float* __restrict__ in,
                             __bf16* __restrict__ out, int n) {
  for (int i = blockIdx.x * blockDim.x + threadIdx.x; i < n;
       i += gridDim.x * blockDim.x)
    out[i] = f2bf(in[i]);
}

// ---------------------------------------------------------------------------
// C[M,N] = A[M,K](bf16) * W[N,K]^T(bf16) + bias[N]   (NT GEMM)
// Block: 256 threads = 8 waves; block tile 128x128; K-step 32.
// Staging via global_load_async_to_lds_b128.
// ---------------------------------------------------------------------------
template <bool OUT_F32>
__global__ __launch_bounds__(256) void gemm_xt(
    const __bf16* __restrict__ A, const __bf16* __restrict__ Bw,
    const float* __restrict__ bias, void* __restrict__ Cout,
    int M, int N, int K) {
  __shared__ __bf16 As[128][40];  // +8 pad: 80B row stride, conflict-free frags
  __shared__ __bf16 Bs[128][40];

  const int tid  = threadIdx.x;
  const int lane = tid & 31;
  const int w    = tid >> 5;
  const int wr   = w & 3;
  const int wc   = w >> 2;
  const int m0   = blockIdx.y * 128;
  const int n0   = blockIdx.x * 128;

  v8f acc[2][4] = {};

  const int sr = tid >> 1;        // staging row 0..127
  const int sc = (tid & 1) * 16;  // staging col 0 or 16

  for (int k0 = 0; k0 < K; k0 += 32) {
    __syncthreads();
    const __bf16* ga = A  + (size_t)(m0 + sr) * K + k0 + sc;
    const __bf16* gb = Bw + (size_t)(n0 + sr) * K + k0 + sc;
    async_ld_b128(&As[sr][sc],     ga);
    async_ld_b128(&As[sr][sc + 8], ga + 8);
    async_ld_b128(&Bs[sr][sc],     gb);
    async_ld_b128(&Bs[sr][sc + 8], gb + 8);
    if (k0 + 32 < K) {                       // global_prefetch_b8 next K-step
      __builtin_prefetch(ga + 32);
      __builtin_prefetch(gb + 32);
    }
    wait_async0();
    __syncthreads();

    v16bf af[2], bfr[4];
    for (int i = 0; i < 2; ++i)
      af[i] = load_frag(&As[wr * 32 + i * 16][0], 40, lane);
    for (int j = 0; j < 4; ++j)
      bfr[j] = load_frag(&Bs[wc * 64 + j * 16][0], 40, lane);
    for (int i = 0; i < 2; ++i)
      for (int j = 0; j < 4; ++j)
        acc[i][j] = wmma_bf16(af[i], bfr[j], acc[i][j]);
  }

  // Epilogue: C layout -> lane holds column n=lane%16, rows m=8*(lane/16)+v
  const int nl = lane & 15;
  const int mh = (lane >> 4) * 8;
  for (int j = 0; j < 4; ++j) {
    const int gn = n0 + wc * 64 + j * 16 + nl;
    const float bv = bias[gn];
    for (int i = 0; i < 2; ++i) {
      for (int v = 0; v < 8; ++v) {
        const int gm = m0 + wr * 32 + i * 16 + mh + v;
        const float val = acc[i][j][v] + bv;
        if constexpr (OUT_F32)
          ((float*)Cout)[(size_t)gm * N + gn] = val;
        else
          ((__bf16*)Cout)[(size_t)gm * N + gn] = f2bf(val);
      }
    }
  }
}

// ---------------------------------------------------------------------------
// Flash attention: grid = (SEQ/64, BATCH*N_HEADS), block = 128 (4 waves).
// Each wave owns one 16-row Q tile; iterates over keys in 32-key tiles with
// online softmax.  K + bias tiles staged with async loads; V staged
// transposed on the VALU path; P bounced through LDS (C- -> A-layout).
// ---------------------------------------------------------------------------
__global__ __launch_bounds__(128) void attn_flash(
    const __bf16* __restrict__ Q, const __bf16* __restrict__ Km,
    const __bf16* __restrict__ Vm, const float* __restrict__ bias_mask,
    const float* __restrict__ head_scale, __bf16* __restrict__ O) {
  __shared__ __bf16 Kt[32][136];     // [k][d], pad -> 272B stride
  __shared__ __bf16 Vt[128][40];     // transposed: [d][k], pad -> 80B stride
  __shared__ __bf16 Pl[4][16][40];   // per-wave P tile [q][k], pad
  __shared__ float  Bl[64][36];      // bias tile [q(64)][k(32)], pad -> 144B

  const int tid  = threadIdx.x;
  const int lane = tid & 31;
  const int w    = tid >> 5;
  const int bh   = blockIdx.y;
  const int b    = bh / N_HEADS;
  const int h    = bh % N_HEADS;
  const int qbase = blockIdx.x * 64;
  const int q0    = qbase + w * 16;

  const float key_scale = 0.08838834764831845f;  // 1/sqrt(128)
  const float hs = head_scale[h];

  const int mrow = lane & 15;        // fragment row / column index
  const int kb   = (lane >> 4) << 3; // fragment K-chunk base
  const int mh   = (lane >> 4) * 8;  // C-tile row base for this lane
  const int nl   = lane & 15;

  // Q fragments: 16 rows x 128 d -> 4 A-operands, resident in VGPRs
  v16bf aq[4];
  {
    const __bf16* qp =
        Q + (size_t)(b * SEQ + q0 + mrow) * D_MODEL + h * D_KEY;
    for (int j = 0; j < 4; ++j) {
      Frag16 f;
      f.h[0] = *(const v8bf*)(qp + j * 32 + kb);
      f.h[1] = *(const v8bf*)(qp + j * 32 + 16 + kb);
      aq[j] = f.v;
    }
  }

  v8f oacc[8] = {};                 // 16 q-rows x 128 d accumulator
  float mrw[8], lrw[8];
  for (int v = 0; v < 8; ++v) { mrw[v] = -1e30f; lrw[v] = 0.f; }

  // Per-thread staging coordinates
  const int kr = tid >> 2;          // 0..31  (K-tile row)
  const int kc = (tid & 3) * 32;    // 0,32,64,96
  const int br = tid >> 1;          // 0..63  (bias row)
  const int bc = (tid & 1) * 16;    // 0 or 16 (bias col)

  for (int kt = 0; kt < SEQ / 32; ++kt) {
    const int k0 = kt * 32;
    __syncthreads();  // protect previous iteration's LDS reads

    // --- async stage K tile (row-major bf16) ---
    const __bf16* kp =
        Km + (size_t)(b * SEQ + k0 + kr) * D_MODEL + h * D_KEY + kc;
    for (int u = 0; u < 4; ++u)
      async_ld_b128(&Kt[kr][kc + u * 8], kp + u * 8);

    // --- async stage bias tile (64 q rows x 32 k, fp32, coalesced) ---
    const float* bmp =
        bias_mask + ((size_t)b * SEQ + qbase + br) * SEQ + k0 + bc;
    for (int u = 0; u < 4; ++u)
      async_ld_b128(&Bl[br][bc + u * 4], bmp + u * 4);

    // --- stage V tile transposed (VALU path: needs d<->k transpose) ---
    for (int e = tid; e < 32 * 32; e += 128) {
      const int rr = e >> 5;
      const int cq = (e & 31) * 4;
      const __bf16* vp =
          Vm + (size_t)(b * SEQ + k0 + rr) * D_MODEL + h * D_KEY + cq;
      __bf16 t0 = vp[0], t1 = vp[1], t2 = vp[2], t3 = vp[3];
      Vt[cq + 0][rr] = t0; Vt[cq + 1][rr] = t1;
      Vt[cq + 2][rr] = t2; Vt[cq + 3][rr] = t3;
    }

    if (kt + 1 < SEQ / 32) {  // prefetch next tiles into L2/WGP$
      __builtin_prefetch(kp + (size_t)32 * D_MODEL);
      __builtin_prefetch(Vm + (size_t)(b * SEQ + k0 + 32 + kr) * D_MODEL +
                         h * D_KEY + kc);
    }
    wait_async0();
    __syncthreads();

    // --- scores: preload all 8 B-fragments, then 8 back-to-back WMMAs ---
    v16bf bk[2][4];
    for (int t = 0; t < 2; ++t) {
      const __bf16* kpp = &Kt[t * 16 + mrow][0];  // B col n = K row n over d
      for (int j = 0; j < 4; ++j) {
        Frag16 f;
        f.h[0] = *(const v8bf*)(kpp + j * 32 + kb);
        f.h[1] = *(const v8bf*)(kpp + j * 32 + 16 + kb);
        bk[t][j] = f.v;
      }
    }
    v8f st[2];
    for (int t = 0; t < 2; ++t) {
      v8f s = {};
      for (int j = 0; j < 4; ++j) s = wmma_bf16(aq[j], bk[t][j], s);
      st[t] = s;
    }

    // --- scale + bias from LDS tile ---
    float p[2][8];
    for (int t = 0; t < 2; ++t)
      for (int v = 0; v < 8; ++v)
        p[t][v] = st[t][v] * key_scale +
                  Bl[w * 16 + mh + v][t * 16 + nl] * hs;

    // --- online softmax (xor-reduce within 16-lane halves) ---
    for (int v = 0; v < 8; ++v) {
      float rx = fmaxf(p[0][v], p[1][v]);
      rx = fmaxf(rx, __shfl_xor(rx, 1, 32));
      rx = fmaxf(rx, __shfl_xor(rx, 2, 32));
      rx = fmaxf(rx, __shfl_xor(rx, 4, 32));
      rx = fmaxf(rx, __shfl_xor(rx, 8, 32));
      const float nm  = fmaxf(mrw[v], rx);
      const float fsc = __expf(mrw[v] - nm);
      const float e0  = __expf(p[0][v] - nm);
      const float e1  = __expf(p[1][v] - nm);
      p[0][v] = e0; p[1][v] = e1;
      float rs = e0 + e1;
      rs += __shfl_xor(rs, 1, 32);
      rs += __shfl_xor(rs, 2, 32);
      rs += __shfl_xor(rs, 4, 32);
      rs += __shfl_xor(rs, 8, 32);
      lrw[v] = lrw[v] * fsc + rs;
      mrw[v] = nm;
      for (int i = 0; i < 8; ++i) oacc[i][v] *= fsc;
    }

    // --- P: C-layout -> LDS -> A-layout (16x32 per wave) ---
    for (int t = 0; t < 2; ++t)
      for (int v = 0; v < 8; ++v)
        Pl[w][mh + v][t * 16 + nl] = f2bf(p[t][v]);
    __syncthreads();

    Frag16 ap;
    {
      const __bf16* pp = &Pl[w][mrow][0];
      ap.h[0] = *(const v8bf*)(pp + kb);
      ap.h[1] = *(const v8bf*)(pp + 16 + kb);
    }
    // --- O += P x V: preload 8 B-fragments, then 8 WMMAs ---
    v16bf bv[8];
    for (int i = 0; i < 8; ++i) {
      Frag16 f;
      const __bf16* vpp = &Vt[i * 16 + mrow][0];
      f.h[0] = *(const v8bf*)(vpp + kb);
      f.h[1] = *(const v8bf*)(vpp + 16 + kb);
      bv[i] = f.v;
    }
    for (int i = 0; i < 8; ++i) oacc[i] = wmma_bf16(ap.v, bv[i], oacc[i]);
  }

  // Normalize and store bf16 attention output [B*S, D_MODEL]
  float inv[8];
  for (int v = 0; v < 8; ++v) inv[v] = 1.0f / lrw[v];
  for (int i = 0; i < 8; ++i) {
    const int gn = h * D_KEY + i * 16 + nl;
    for (int v = 0; v < 8; ++v) {
      const int gm = b * SEQ + q0 + mh + v;
      O[(size_t)gm * D_MODEL + gn] = f2bf(oacc[i][v] * inv[v]);
    }
  }
}

// ---------------------------------------------------------------------------
// Host-side orchestration
// ---------------------------------------------------------------------------
extern "C" void kernel_launch(void* const* d_in, const int* in_sizes, int n_in,
                              void* d_out, int out_size, void* d_ws,
                              size_t ws_size, hipStream_t stream) {
  (void)in_sizes; (void)n_in; (void)out_size; (void)ws_size;

  const float* states     = (const float*)d_in[0];
  const float* bias_mask  = (const float*)d_in[1];
  const float* head_scale = (const float*)d_in[2];
  const float* qw = (const float*)d_in[3];
  const float* qb = (const float*)d_in[4];
  const float* kw = (const float*)d_in[5];
  const float* kbi = (const float*)d_in[6];
  const float* vw = (const float*)d_in[7];
  const float* vb = (const float*)d_in[8];
  const float* ow = (const float*)d_in[9];
  const float* ob = (const float*)d_in[10];
  float* out = (float*)d_out;

  // Workspace layout (bf16 everywhere): ~117 MB total
  char* ws = (char*)d_ws;
  const size_t szX = (size_t)MTOT * D_MODEL;       // 8,388,608 elems
  const size_t szW = (size_t)D_MODEL * D_MODEL;    // 4,194,304 elems
  __bf16* Xbf = (__bf16*)ws;              ws += szX * 2;
  __bf16* Wqb = (__bf16*)ws;              ws += szW * 2;
  __bf16* Wkb = (__bf16*)ws;              ws += szW * 2;
  __bf16* Wvb = (__bf16*)ws;              ws += szW * 2;
  __bf16* Wob = (__bf16*)ws;              ws += szW * 2;
  __bf16* Qb  = (__bf16*)ws;              ws += szX * 2;
  __bf16* Kb  = (__bf16*)ws;              ws += szX * 2;
  __bf16* Vb  = (__bf16*)ws;              ws += szX * 2;
  __bf16* Ab  = (__bf16*)ws;              ws += szX * 2;

  // 1) fp32 -> bf16 conversions
  cvt_f32_bf16<<<2048, 256, 0, stream>>>(states, Xbf, (int)szX);
  cvt_f32_bf16<<<2048, 256, 0, stream>>>(qw, Wqb, (int)szW);
  cvt_f32_bf16<<<2048, 256, 0, stream>>>(kw, Wkb, (int)szW);
  cvt_f32_bf16<<<2048, 256, 0, stream>>>(vw, Wvb, (int)szW);
  cvt_f32_bf16<<<2048, 256, 0, stream>>>(ow, Wob, (int)szW);

  // 2) Q/K/V projections: [4096,2048] = X * W^T + b  (bf16 out)
  dim3 ggrid(D_MODEL / 128, MTOT / 128);  // (16, 32)
  gemm_xt<false><<<ggrid, 256, 0, stream>>>(Xbf, Wqb, qb, (void*)Qb,
                                            MTOT, D_MODEL, D_MODEL);
  gemm_xt<false><<<ggrid, 256, 0, stream>>>(Xbf, Wkb, kbi, (void*)Kb,
                                            MTOT, D_MODEL, D_MODEL);
  gemm_xt<false><<<ggrid, 256, 0, stream>>>(Xbf, Wvb, vb, (void*)Vb,
                                            MTOT, D_MODEL, D_MODEL);

  // 3) Flash attention -> bf16 [4096, 2048]
  dim3 agrid(SEQ / 64, BATCH * N_HEADS);  // (32, 32)
  attn_flash<<<agrid, 128, 0, stream>>>(Qb, Kb, Vb, bias_mask, head_scale, Ab);

  // 4) Output projection, fp32 result
  gemm_xt<true><<<ggrid, 256, 0, stream>>>(Ab, Wob, ob, (void*)out,
                                           MTOT, D_MODEL, D_MODEL);
}
